// GemmaAttention_58643483459800
// MI455X (gfx1250) — matrix operand order repaired and checked by reading, not compiled, not run on hardware
//
#include <hip/hip_runtime.h>
#include <stdint.h>
#include <string.h>

// ---- problem constants (match reference) ----
#define B_    2
#define S_    2048
#define D_    2048
#define NH_   16
#define NKV_  8
#define HD_   128
#define E_    4096            // (NH + 2*NKV) * HD
#define MTOT_ 4096            // B * S
#define SCALE_ 0.08838834764831845f   // HD^-0.5
#define LOG2E_ 1.4426950408889634f

typedef __attribute__((ext_vector_type(16))) __bf16 v16bf;
typedef __attribute__((ext_vector_type(8)))  float  v8f;

union FragB { v16bf v; unsigned u[8]; };

__device__ inline unsigned short f2bf(float f) {
  unsigned u = __builtin_bit_cast(unsigned, f);
  u += 0x7FFFu + ((u >> 16) & 1u);      // round-to-nearest-even
  return (unsigned short)(u >> 16);
}

__device__ inline v8f wmma_bf16(const FragB& a, const FragB& b, v8f c) {
  return __builtin_amdgcn_wmma_f32_16x16x32_bf16(false, a.v, false, b.v,
                                                 (short)0, c, false, false);
}

// A-fragment K offset for pair j (2 bf16 per VGPR), lane-half lh.
__device__ inline int akoff(int j, int lh) {
  return 16 * lh + 2 * j;
}

// ---- CDNA5 async global->LDS copy (ASYNCcnt path, ISA 10.x / 15.18) ----
__device__ inline unsigned lds_off_u32(const void* p) {
  return (unsigned)(size_t)(__attribute__((address_space(3))) const void*)p;
}
__device__ inline void async_copy_b128(const void* g, const void* lds) {
  asm volatile("global_load_async_to_lds_b128 %0, %1, off"
               :: "v"(lds_off_u32(lds)), "v"(g) : "memory");
}
__device__ inline void wait_async0() {
  asm volatile("s_wait_asynccnt 0" ::: "memory");
}

// ---------------- f32 -> bf16 conversion ----------------
__global__ void cvt_bf16_kernel(const float* __restrict__ in,
                                unsigned short* __restrict__ out, int n) {
  int i = blockIdx.x * blockDim.x + threadIdx.x;
  int stride = gridDim.x * blockDim.x;
  for (; i < n; i += stride) out[i] = f2bf(in[i]);
}

// ---------------- shared GEMM tile core (128x128, K=2048) ----------------
struct __align__(16) GemmSmem {
  union {
    unsigned short ab[2][2][128][40];   // [buf][A/B][row][k] bf16, double-buffered
    float c[128][132];                  // C staging (reused after K-loop)
  };
};

// Issue async copies of one 128x32 A tile and one 128x32 B tile into buf.
__device__ inline void load_tiles_async(const unsigned short* __restrict__ A,
                                        const unsigned short* __restrict__ Bm,
                                        int mbase, int nbase, int ks,
                                        GemmSmem& sm, int buf) {
  const int tid  = threadIdx.x;
  const int row  = tid >> 1;            // 0..127
  const int half = tid & 1;             // cols half*16 .. +15
  const unsigned short* ga = A  + (size_t)(mbase + row) * D_ + ks + half * 16;
  const unsigned short* gb = Bm + (size_t)(nbase + row) * D_ + ks + half * 16;
  async_copy_b128(ga,     &sm.ab[buf][0][row][half * 16]);
  async_copy_b128(ga + 8, &sm.ab[buf][0][row][half * 16 + 8]);
  async_copy_b128(gb,     &sm.ab[buf][1][row][half * 16]);
  async_copy_b128(gb + 8, &sm.ab[buf][1][row][half * 16 + 8]);
}

// C = A(128 rows from mbase) x B^T(128 rows from nbase); A,B row-major [*,2048]
// bf16. 256 threads = 8 waves in a 4(m) x 2(n) grid; each wave owns 32x64
// (8 accumulating WMMAs per 32-wide K step). Result staged into sm.c.
__device__ inline void gemm128x128_bf16(const unsigned short* __restrict__ A,
                                        const unsigned short* __restrict__ Bm,
                                        int mbase, int nbase, GemmSmem& sm) {
  const int tid  = threadIdx.x;
  const int lane = tid & 31;
  const int wid  = tid >> 5;
  const int wm = wid & 3;               // 4 row groups of 32
  const int wn = wid >> 2;              // 2 col groups of 64
  const int lh = lane >> 4, ln = lane & 15;

  v8f acc[2][4];
#pragma unroll
  for (int i = 0; i < 2; ++i)
#pragma unroll
    for (int t = 0; t < 4; ++t)
#pragma unroll
      for (int r = 0; r < 8; ++r) acc[i][t][r] = 0.0f;

  load_tiles_async(A, Bm, mbase, nbase, 0, sm, 0);
  wait_async0();
  __syncthreads();

  const int steps = D_ / 32;
  for (int s32 = 0; s32 < steps; ++s32) {
    const int cur = s32 & 1;
    if (s32 + 1 < steps)                // prefetch next tile while computing
      load_tiles_async(A, Bm, mbase, nbase, (s32 + 1) * 32, sm, cur ^ 1);

    FragB fa[2];
#pragma unroll
    for (int i = 0; i < 2; ++i)
#pragma unroll
      for (int j = 0; j < 8; ++j)
        fa[i].u[j] =
            *(const unsigned*)&sm.ab[cur][0][wm * 32 + i * 16 + ln][akoff(j, lh)];

#pragma unroll
    for (int t = 0; t < 4; ++t) {
      FragB fb;
#pragma unroll
      for (int j = 0; j < 8; ++j)
        fb.u[j] =
            *(const unsigned*)&sm.ab[cur][1][wn * 64 + t * 16 + ln][16 * lh + 2 * j];
      acc[0][t] = wmma_bf16(fa[0], fb, acc[0][t]);
      acc[1][t] = wmma_bf16(fa[1], fb, acc[1][t]);
    }

    wait_async0();
    __syncthreads();
  }

  // stage C (union with ab buffers; all reads drained at last barrier)
#pragma unroll
  for (int i = 0; i < 2; ++i)
#pragma unroll
    for (int t = 0; t < 4; ++t)
#pragma unroll
      for (int r = 0; r < 8; ++r)
        sm.c[wm * 32 + i * 16 + r + 8 * lh][wn * 64 + t * 16 + ln] = acc[i][t][r];
  __syncthreads();
}

// ---------------- kernel 2: QKV GEMM + bias + RoPE + scatter ----------------
__global__ __launch_bounds__(256)
void qkv_rope_kernel(const unsigned short* __restrict__ xh,
                     const unsigned short* __restrict__ wqh,
                     const float* __restrict__ bqkv,
                     const float* __restrict__ fc,          // [S][HD/2][2]
                     unsigned short* __restrict__ qbuf,     // [B][NH][S][HD]
                     unsigned short* __restrict__ kbuf,     // [B][NKV][S][HD]
                     unsigned short* __restrict__ vbuf) {   // [B][NKV][S][HD]
  __shared__ GemmSmem sm;
  const int nbase = blockIdx.x * 128;
  const int mbase = blockIdx.y * 128;
  gemm128x128_bf16(xh, wqh, mbase, nbase, sm);

  // epilogue: 128 rows x 64 RoPE pairs per tile
  for (int p = threadIdx.x; p < 8192; p += 256) {
    const int r  = p >> 6;         // tile row
    const int pi = p & 63;         // pair index within tile cols
    const int gr = mbase + r;      // global row = b*S + s
    const int b  = gr >> 11;
    const int s  = gr & (S_ - 1);
    const int n0 = nbase + pi * 2;
    float x0 = sm.c[r][pi * 2]     + bqkv[n0];
    float x1 = sm.c[r][pi * 2 + 1] + bqkv[n0 + 1];

    if (n0 < NH_ * HD_) {                       // Q + RoPE
      const int h = n0 >> 7, d = n0 & 127, i = d >> 1;
      const float c  = fc[(s * 64 + i) * 2];
      const float sn = fc[(s * 64 + i) * 2 + 1];
      const float y0 = x0 * c - x1 * sn;
      const float y1 = x0 * sn + x1 * c;
      const size_t o = ((size_t)(b * NH_ + h) * S_ + s) * HD_ + d;
      qbuf[o] = f2bf(y0); qbuf[o + 1] = f2bf(y1);
    } else if (n0 < (NH_ + NKV_) * HD_) {       // K + RoPE
      const int m = n0 - NH_ * HD_;
      const int hk = m >> 7, d = m & 127, i = d >> 1;
      const float c  = fc[(s * 64 + i) * 2];
      const float sn = fc[(s * 64 + i) * 2 + 1];
      const float y0 = x0 * c - x1 * sn;
      const float y1 = x0 * sn + x1 * c;
      const size_t o = ((size_t)(b * NKV_ + hk) * S_ + s) * HD_ + d;
      kbuf[o] = f2bf(y0); kbuf[o + 1] = f2bf(y1);
    } else {                                    // V
      const int m = n0 - (NH_ + NKV_) * HD_;
      const int hk = m >> 7, d = m & 127;
      const size_t o = ((size_t)(b * NKV_ + hk) * S_ + s) * HD_ + d;
      vbuf[o] = f2bf(x0); vbuf[o + 1] = f2bf(x1);
    }
  }
}

// ---------------- kernel 3: causal flash attention ----------------
__global__ __launch_bounds__(128)
void attn_kernel(const unsigned short* __restrict__ qb,
                 const unsigned short* __restrict__ kb,
                 const unsigned short* __restrict__ vb,
                 unsigned short* __restrict__ ao) {        // [B][S][NH*HD]
  __shared__ __align__(16) unsigned short Kt[32][136];     // [kv][d]
  __shared__ __align__(16) unsigned short Vt[128][40];     // [d][kv] transposed
  __shared__ __align__(16) unsigned short Pl[4][16][40];   // per-wave P tile

  const int tid  = threadIdx.x;
  const int lane = tid & 31;
  const int wid  = tid >> 5;
  const int lh = lane >> 4, ln = lane & 15;

  const int qtile = blockIdx.x;
  const int bh    = blockIdx.y;
  const int b  = bh >> 4;
  const int h  = bh & 15;
  const int hk = h >> 1;                      // GQA: 2 q-heads per kv-head
  const int qbase = qtile * 64 + wid * 16;    // this wave's 16 q-rows

  // Q fragments: 16 rows x 128 head-dim = 4 A-fragments, loaded once.
  FragB qf[4];
  {
    const unsigned short* qrow =
        qb + ((size_t)(b * NH_ + h) * S_ + (qbase + ln)) * HD_;
#pragma unroll
    for (int kc = 0; kc < 4; ++kc)
#pragma unroll
      for (int j = 0; j < 8; ++j)
        qf[kc].u[j] = *(const unsigned*)(qrow + kc * 32 + akoff(j, lh));
  }

  v8f o[8];
#pragma unroll
  for (int dj = 0; dj < 8; ++dj)
#pragma unroll
    for (int r = 0; r < 8; ++r) o[dj][r] = 0.0f;

  float mrow[8], lrow[8];
#pragma unroll
  for (int r = 0; r < 8; ++r) { mrow[r] = -1e30f; lrow[r] = 0.0f; }

  const unsigned short* kh = kb + (size_t)(b * NKV_ + hk) * S_ * HD_;
  const unsigned short* vh = vb + (size_t)(b * NKV_ + hk) * S_ * HD_;

  const int nch = qtile * 2 + 2;              // kv chunks of 32 (causal bound)
  const int lrow_g = tid >> 2, lseg = tid & 3;

  for (int c = 0; c < nch; ++c) {
    const int kv0 = c * 32;
    __syncthreads();
    {
      // K chunk: 32 x 128 bf16 via async global->LDS (ASYNCcnt)
      const unsigned short* ks = kh + (size_t)(kv0 + lrow_g) * HD_ + lseg * 32;
#pragma unroll
      for (int g = 0; g < 4; ++g)
        async_copy_b128(ks + g * 8, &Kt[lrow_g][lseg * 32 + g * 8]);
      // V chunk transposed into Vt[d][kv] (register transpose)
      const unsigned* vs =
          (const unsigned*)(vh + (size_t)(kv0 + lrow_g) * HD_ + lseg * 32);
#pragma unroll
      for (int g = 0; g < 16; ++g) {
        unsigned w = vs[g];
        Vt[lseg * 32 + 2 * g][lrow_g]     = (unsigned short)(w & 0xFFFFu);
        Vt[lseg * 32 + 2 * g + 1][lrow_g] = (unsigned short)(w >> 16);
      }
    }
    wait_async0();
    __syncthreads();

    // ---- scores: S = Q x K^T, two 16x16 tiles over 32 kv columns ----
    v8f sc[2];
#pragma unroll
    for (int t = 0; t < 2; ++t) {
      v8f a;
#pragma unroll
      for (int r = 0; r < 8; ++r) a[r] = 0.0f;
#pragma unroll
      for (int kc = 0; kc < 4; ++kc) {
        FragB fb;
#pragma unroll
        for (int j = 0; j < 8; ++j)
          fb.u[j] = *(const unsigned*)&Kt[t * 16 + ln][kc * 32 + 16 * lh + 2 * j];
        a = wmma_bf16(qf[kc], fb, a);
      }
      sc[t] = a;
    }

    // ---- scale + causal mask ----
#pragma unroll
    for (int t = 0; t < 2; ++t) {
      const int kvp = kv0 + t * 16 + ln;
#pragma unroll
      for (int r = 0; r < 8; ++r) {
        const int qp = qbase + r + 8 * lh;
        const float v = sc[t][r] * SCALE_;
        sc[t][r] = (kvp <= qp) ? v : -1e30f;
      }
    }

    // ---- online softmax (rows live in half-wave lane groups of 16) ----
#pragma unroll
    for (int r = 0; r < 8; ++r) {
      float cm = fmaxf(sc[0][r], sc[1][r]);
#pragma unroll
      for (int off = 8; off >= 1; off >>= 1) cm = fmaxf(cm, __shfl_xor(cm, off, 16));
      const float mn = fmaxf(mrow[r], cm);
      const float al = exp2f((mrow[r] - mn) * LOG2E_);
      const float e0 = exp2f((sc[0][r] - mn) * LOG2E_);
      const float e1 = exp2f((sc[1][r] - mn) * LOG2E_);
      float rs = e0 + e1;
#pragma unroll
      for (int off = 8; off >= 1; off >>= 1) rs += __shfl_xor(rs, off, 16);
      lrow[r] = lrow[r] * al + rs;
      mrow[r] = mn;
#pragma unroll
      for (int dj = 0; dj < 8; ++dj) o[dj][r] *= al;
      // stash probs (C layout row = r + 8*lh) for A-fragment re-layout
      Pl[wid][r + 8 * lh][ln]      = f2bf(e0);
      Pl[wid][r + 8 * lh][16 + ln] = f2bf(e1);
    }
    __syncthreads();

    // ---- O += P x V ----
    FragB pf;
#pragma unroll
    for (int j = 0; j < 8; ++j)
      pf.u[j] = *(const unsigned*)&Pl[wid][ln][akoff(j, lh)];
#pragma unroll
    for (int dj = 0; dj < 8; ++dj) {
      FragB fv;
#pragma unroll
      for (int j = 0; j < 8; ++j)
        fv.u[j] = *(const unsigned*)&Vt[dj * 16 + ln][16 * lh + 2 * j];
      o[dj] = wmma_bf16(pf, fv, o[dj]);
    }
  }

  // ---- finalize: O /= l, store bf16 to [B][S][NH*HD] ----
#pragma unroll
  for (int r = 0; r < 8; ++r) {
    const float inv = 1.0f / lrow[r];
    const int qp = qbase + r + 8 * lh;
#pragma unroll
    for (int dj = 0; dj < 8; ++dj) {
      const size_t off =
          ((size_t)b * S_ + qp) * (NH_ * HD_) + h * HD_ + dj * 16 + ln;
      ao[off] = f2bf(o[dj][r] * inv);
    }
  }
}

// ---------------- kernel 4: output projection + bias ----------------
__global__ __launch_bounds__(256)
void oproj_kernel(const unsigned short* __restrict__ aoh,
                  const unsigned short* __restrict__ woh,
                  const float* __restrict__ bo,
                  float* __restrict__ out) {
  __shared__ GemmSmem sm;
  const int nbase = blockIdx.x * 128;
  const int mbase = blockIdx.y * 128;
  gemm128x128_bf16(aoh, woh, mbase, nbase, sm);

  for (int e = threadIdx.x; e < 16384; e += 256) {
    const int r = e >> 7, col = e & 127;
    const int gr = mbase + r;
    const int n  = nbase + col;
    out[(size_t)gr * D_ + n] = sm.c[r][col] + bo[n];
  }
}

// ---------------- host launcher ----------------
extern "C" void kernel_launch(void* const* d_in, const int* in_sizes, int n_in,
                              void* d_out, int out_size, void* d_ws, size_t ws_size,
                              hipStream_t stream) {
  (void)in_sizes; (void)n_in; (void)out_size; (void)ws_size;
  const float* x    = (const float*)d_in[0];
  const float* fc   = (const float*)d_in[1];
  // d_in[2] = mask: causal, applied analytically in-kernel.
  const float* Wqkv = (const float*)d_in[3];
  const float* bqkv = (const float*)d_in[4];
  const float* Wo   = (const float*)d_in[5];
  const float* bo   = (const float*)d_in[6];
  float* out = (float*)d_out;

  unsigned short* xh   = (unsigned short*)d_ws;
  unsigned short* wqh  = xh   + (size_t)MTOT_ * D_;
  unsigned short* woh  = wqh  + (size_t)E_ * D_;
  unsigned short* qbuf = woh  + (size_t)D_ * D_;
  unsigned short* kbuf = qbuf + (size_t)B_ * NH_  * S_ * HD_;
  unsigned short* vbuf = kbuf + (size_t)B_ * NKV_ * S_ * HD_;
  unsigned short* aob  = vbuf + (size_t)B_ * NKV_ * S_ * HD_;
  // total workspace: ~92.3 MB

  cvt_bf16_kernel<<<2048, 256, 0, stream>>>(x,    xh,  MTOT_ * D_);
  cvt_bf16_kernel<<<2048, 256, 0, stream>>>(Wqkv, wqh, E_ * D_);
  cvt_bf16_kernel<<<1024, 256, 0, stream>>>(Wo,   woh, D_ * D_);

  qkv_rope_kernel<<<dim3(E_ / 128, MTOT_ / 128), 256, 0, stream>>>(
      xh, wqh, bqkv, fc, qbuf, kbuf, vbuf);

  attn_kernel<<<dim3(S_ / 64, B_ * NH_), 128, 0, stream>>>(qbuf, kbuf, vbuf, aob);

  oproj_kernel<<<dim3(D_ / 128, MTOT_ / 128), 256, 0, stream>>>(aob, woh, bo, out);
}
